// LCSA_52338471469594
// MI455X (gfx1250) — matrix operand order repaired and checked by reading, not compiled, not run on hardware
//
#include <hip/hip_runtime.h>
#include <hip/hip_bf16.h>
#include <stdint.h>

typedef __attribute__((ext_vector_type(16))) __bf16   v16bf;
typedef __attribute__((ext_vector_type(8)))  float    v8f;
typedef __attribute__((ext_vector_type(8)))  uint32_t v8u;

#define B_   2
#define L_   1024
#define PAD_ 120            // (KERNEL-1)*max(dil) = 15*8
#define LP_  (L_ + PAD_)    // front-padded length (forward mode only looks back)
#define DM_  512
#define H_   8
#define HW_  64
#define K_   16
#define MT_  2              // l-tiles (of 16) per block: M-blocking for B-fragment reuse
#define ML_  (MT_ * 16)     // 32 l rows per block

__device__ __forceinline__ uint16_t f2bf(float f) {
  uint32_t u = __float_as_uint(f);
  uint32_t r = u + 0x7FFFu + ((u >> 16) & 1u);   // round-to-nearest-even
  return (uint16_t)(r >> 16);
}
__device__ __forceinline__ float bf2f(uint16_t h) {
  return __uint_as_float(((uint32_t)h) << 16);
}

// ---------------- conversion kernels ----------------
__global__ void cvt_bf16(const float* __restrict__ src, uint16_t* __restrict__ dst, int n) {
  int i = blockIdx.x * blockDim.x + threadIdx.x;
  if (i < n) dst[i] = f2bf(src[i]);
}

__global__ void build_xpad(const float* __restrict__ x, uint16_t* __restrict__ xpad) {
  int i = blockIdx.x * blockDim.x + threadIdx.x;
  const int total = B_ * LP_ * DM_;
  if (i >= total) return;
  int d    = i % DM_;
  int rest = i / DM_;
  int p    = rest % LP_;
  int b    = rest / LP_;
  float f = (p < PAD_) ? 0.f : x[((size_t)b * L_ + (p - PAD_)) * DM_ + d];
  xpad[i] = f2bf(f);
}

// ---------------- fused QKV projection + attention (per b,h,32-l tile) ----------------
__global__ __launch_bounds__(256) void lcsa_qkv_attn(
    const uint16_t* __restrict__ xpad,
    const uint16_t* __restrict__ wq, const uint16_t* __restrict__ wk,
    const uint16_t* __restrict__ wv,
    const float* __restrict__ bq, const float* __restrict__ bk,
    const float* __restrict__ bv,
    uint16_t* __restrict__ attn_ws)
{
  __shared__ float    lds_q[ML_][HW_];          // 8 KB
  __shared__ uint16_t lds_k[K_][ML_][HW_];      // 64 KB
  __shared__ float    lds_logit[ML_][K_];       // 2 KB
  __shared__ float    lds_score[ML_][K_];       // 2 KB
  __shared__ float    lds_attn[ML_][HW_];       // 8 KB

  const int l0  = blockIdx.x * ML_;
  const int h   = blockIdx.y;
  const int b   = blockIdx.z;
  const int dil = (h < 2) ? 1 : (h < 4) ? 2 : (h < 6) ? 4 : 8;

  const int tid  = threadIdx.x;
  const int wave = tid >> 5;
  const int lane = tid & 31;
  const int n    = lane & 15;   // column / row-within-half
  const int hi   = lane >> 4;   // half-wave select
  const int akb  = hi * 8;      // A-frag K base (also C-frag row base)
  const int bkb  = hi * 16;     // unused; B-frag K base is akb, same as A

  const uint16_t* xb = xpad + (size_t)b * LP_ * DM_;

  v8f vkeep[2][MT_][4];         // V tiles kept in registers across barriers
  int vkid[2];
  int nv = 0;

  // 33 GEMM jobs of shape (32x512)x(512x64): job0=Q, 1..16=K_k, 17..32=V_k
  for (int job = wave; job < 33; job += 8) {
    int rowbase, kk = 0, kind;
    const uint16_t* Wb;
    const float*    bias;
    if (job == 0) {
      kind = 0;
      rowbase = PAD_ + l0;
      Wb   = wq + (size_t)h * HW_ * DM_;
      bias = bq + h * HW_;
    } else if (job <= 16) {
      kind = 1; kk = job - 1;
      rowbase = PAD_ + l0 + (kk - (K_ - 1)) * dil;
      Wb   = wk + (size_t)((h * K_ + kk) * HW_) * DM_;
      bias = bk + (h * K_ + kk) * HW_;
    } else {
      kind = 2; kk = job - 17;
      rowbase = PAD_ + l0 + (kk - (K_ - 1)) * dil;
      Wb   = wv + (size_t)((h * K_ + kk) * HW_) * DM_;
      bias = bv + (h * K_ + kk) * HW_;
    }

    const uint16_t* arow0 = xb + (size_t)(rowbase + n) * DM_;
    const uint16_t* arow1 = arow0 + 16 * DM_;
    v8f acc[MT_][4] = {};

    for (int kc = 0; kc < 16; ++kc) {             // 512 = 16 chunks of K=32
      v8u au0, au1;
      #pragma unroll
      for (int v = 0; v < 8; ++v) {
        int Kof = kc * 32 + ((v & 4) ? 16 : 0) + akb + 2 * (v & 3);
        au0[v] = *(const uint32_t*)(arow0 + Kof);
        au1[v] = *(const uint32_t*)(arow1 + Kof);
      }
      v16bf a0 = __builtin_bit_cast(v16bf, au0);
      v16bf a1 = __builtin_bit_cast(v16bf, au1);
      #pragma unroll
      for (int ot = 0; ot < 4; ++ot) {            // 64 output cols = 4 N-tiles
        const uint16_t* brow = Wb + (size_t)(ot * 16 + n) * DM_;
        v8u bu;
        #pragma unroll
        for (int v = 0; v < 8; ++v) {
          int Kof = kc * 32 + ((v & 4) ? 16 : 0) + akb + 2 * (v & 3);
          bu[v] = *(const uint32_t*)(brow + Kof);
        }
        v16bf bmat = __builtin_bit_cast(v16bf, bu);
        acc[0][ot] = __builtin_amdgcn_wmma_f32_16x16x32_bf16(
            false, a0, false, bmat, (short)0, acc[0][ot], false, false);
        acc[1][ot] = __builtin_amdgcn_wmma_f32_16x16x32_bf16(
            false, a1, false, bmat, (short)0, acc[1][ot], false, false);
      }
    }

    // bias (in place)
    #pragma unroll
    for (int mt = 0; mt < MT_; ++mt)
      #pragma unroll
      for (int ot = 0; ot < 4; ++ot) {
        float bsv = bias[ot * 16 + n];
        #pragma unroll
        for (int v = 0; v < 8; ++v) acc[mt][ot][v] += bsv;
      }

    if (kind == 0) {
      #pragma unroll
      for (int mt = 0; mt < MT_; ++mt)
        #pragma unroll
        for (int ot = 0; ot < 4; ++ot)
          #pragma unroll
          for (int v = 0; v < 8; ++v)
            lds_q[mt * 16 + v + akb][ot * 16 + n] = acc[mt][ot][v];
    } else if (kind == 1) {
      #pragma unroll
      for (int mt = 0; mt < MT_; ++mt)
        #pragma unroll
        for (int ot = 0; ot < 4; ++ot)
          #pragma unroll
          for (int v = 0; v < 8; ++v)
            lds_k[kk][mt * 16 + v + akb][ot * 16 + n] = f2bf(acc[mt][ot][v]);
    } else {
      vkid[nv] = kk;
      #pragma unroll
      for (int mt = 0; mt < MT_; ++mt)
        #pragma unroll
        for (int ot = 0; ot < 4; ++ot) vkeep[nv][mt][ot] = acc[mt][ot];
      ++nv;
    }
  }

  // zero attn accumulator before first barrier (ML_*HW_ = 2048 floats)
  #pragma unroll
  for (int i = 0; i < (ML_ * HW_) / 256; ++i) {
    int idx = tid + i * 256;
    lds_attn[idx >> 6][idx & 63] = 0.f;
  }
  __syncthreads();

  // logits[l][k] = q[l,:] . k[l,k,:]   (one thread per (l,k) pair; 512 pairs)
  #pragma unroll
  for (int i = 0; i < (ML_ * K_) / 256; ++i) {
    int idx = tid + i * 256;
    int l = idx >> 4, k = idx & 15;
    float s = 0.f;
    #pragma unroll 8
    for (int o = 0; o < HW_; ++o)
      s += lds_q[l][o] * bf2f(lds_k[k][l][o]);
    lds_logit[l][k] = s;
  }
  __syncthreads();

  // softmax over k (row-redundant)
  #pragma unroll
  for (int i = 0; i < (ML_ * K_) / 256; ++i) {
    int idx = tid + i * 256;
    int l = idx >> 4, k = idx & 15;
    float mx = -3.402823466e38f;
    #pragma unroll
    for (int j = 0; j < K_; ++j) mx = fmaxf(mx, lds_logit[l][j]);
    float den = 0.f;
    #pragma unroll
    for (int j = 0; j < K_; ++j) den += __expf(lds_logit[l][j] - mx);
    lds_score[l][k] = __expf(lds_logit[l][k] - mx) / den;
  }
  __syncthreads();

  // attn[l,o] += score[l,k] * v[l,k,o] using register-held V tiles
  for (int i = 0; i < nv; ++i) {
    int kk = vkid[i];
    #pragma unroll
    for (int mt = 0; mt < MT_; ++mt)
      #pragma unroll
      for (int ot = 0; ot < 4; ++ot)
        #pragma unroll
        for (int v = 0; v < 8; ++v) {
          int m = mt * 16 + v + akb;
          atomicAdd(&lds_attn[m][ot * 16 + n], vkeep[i][mt][ot][v] * lds_score[m][kk]);
        }
  }
  __syncthreads();

  // write scaled attn to ws as bf16, layout [b][l][h*64+o] so K-dim matches Wo[d,h,o]
  #pragma unroll
  for (int i = 0; i < (ML_ * HW_) / 256; ++i) {
    int idx = tid + i * 256;
    int m = idx >> 6, o = idx & 63;
    attn_ws[((size_t)(b * L_ + l0 + m)) * DM_ + h * HW_ + o] =
        f2bf(lds_attn[m][o] * 0.125f);   // 1/sqrt(HEAD_W)
  }
}

// ---------------- output projection: out(2048x512) = attn(2048x512) * Wo^T + bo ----------------
__global__ __launch_bounds__(256) void lcsa_outproj(
    const uint16_t* __restrict__ attn_ws, const uint16_t* __restrict__ wo,
    const float* __restrict__ bo, float* __restrict__ out)
{
  const int mtile0 = blockIdx.x * MT_;
  const int tid    = threadIdx.x;
  const int wave   = tid >> 5;
  const int lane   = tid & 31;
  const int n      = lane & 15;
  const int hi     = lane >> 4;
  const int akb    = hi * 8;
  const int bkb    = hi * 16;

  const uint16_t* arow0 = attn_ws + (size_t)(mtile0 * 16 + n) * DM_;
  const uint16_t* arow1 = arow0 + 16 * DM_;
  v8f acc[MT_][4] = {};

  for (int kc = 0; kc < 16; ++kc) {
    v8u au0, au1;
    #pragma unroll
    for (int v = 0; v < 8; ++v) {
      int Kof = kc * 32 + ((v & 4) ? 16 : 0) + akb + 2 * (v & 3);
      au0[v] = *(const uint32_t*)(arow0 + Kof);
      au1[v] = *(const uint32_t*)(arow1 + Kof);
    }
    v16bf a0 = __builtin_bit_cast(v16bf, au0);
    v16bf a1 = __builtin_bit_cast(v16bf, au1);
    #pragma unroll
    for (int j = 0; j < 4; ++j) {
      int d = (wave * 4 + j) * 16 + n;                // output column
      const uint16_t* brow = wo + (size_t)d * DM_;    // Wo[d][ho] contiguous over ho=K
      v8u bu;
      #pragma unroll
      for (int v = 0; v < 8; ++v) {
        int Kof = kc * 32 + ((v & 4) ? 16 : 0) + akb + 2 * (v & 3);
        bu[v] = *(const uint32_t*)(brow + Kof);
      }
      v16bf bmat = __builtin_bit_cast(v16bf, bu);
      acc[0][j] = __builtin_amdgcn_wmma_f32_16x16x32_bf16(
          false, a0, false, bmat, (short)0, acc[0][j], false, false);
      acc[1][j] = __builtin_amdgcn_wmma_f32_16x16x32_bf16(
          false, a1, false, bmat, (short)0, acc[1][j], false, false);
    }
  }

  #pragma unroll
  for (int mt = 0; mt < MT_; ++mt)
    #pragma unroll
    for (int j = 0; j < 4; ++j) {
      int d = (wave * 4 + j) * 16 + n;
      float bias = bo[d];
      #pragma unroll
      for (int v = 0; v < 8; ++v) {
        int R = (mtile0 + mt) * 16 + v + akb;
        out[(size_t)R * DM_ + d] = acc[mt][j][v] + bias;
      }
    }
}

// ---------------- launch ----------------
extern "C" void kernel_launch(void* const* d_in, const int* in_sizes, int n_in,
                              void* d_out, int out_size, void* d_ws, size_t ws_size,
                              hipStream_t stream) {
  (void)in_sizes; (void)n_in; (void)out_size; (void)ws_size;
  const float* x  = (const float*)d_in[0];
  const float* Wq = (const float*)d_in[1];
  const float* bq = (const float*)d_in[2];
  const float* Wk = (const float*)d_in[3];
  const float* bk = (const float*)d_in[4];
  const float* Wv = (const float*)d_in[5];
  const float* bv = (const float*)d_in[6];
  const float* Wo = (const float*)d_in[7];
  const float* bo = (const float*)d_in[8];
  float* out = (float*)d_out;

  char* ws = (char*)d_ws;
  auto align256 = [](size_t v) { return (v + 255) & ~(size_t)255; };
  size_t off = 0;
  uint16_t* xpad  = (uint16_t*)(ws + off); off = align256(off + (size_t)B_ * LP_ * DM_ * 2);
  uint16_t* wqb   = (uint16_t*)(ws + off); off = align256(off + (size_t)H_ * HW_ * DM_ * 2);
  uint16_t* wkb   = (uint16_t*)(ws + off); off = align256(off + (size_t)H_ * K_ * HW_ * DM_ * 2);
  uint16_t* wvb   = (uint16_t*)(ws + off); off = align256(off + (size_t)H_ * K_ * HW_ * DM_ * 2);
  uint16_t* wob   = (uint16_t*)(ws + off); off = align256(off + (size_t)DM_ * DM_ * 2);
  uint16_t* attnb = (uint16_t*)(ws + off); off = align256(off + (size_t)B_ * L_ * DM_ * 2);

  build_xpad<<<(B_ * LP_ * DM_ + 255) / 256, 256, 0, stream>>>(x, xpad);
  cvt_bf16<<<(H_ * HW_ * DM_ + 255) / 256, 256, 0, stream>>>(Wq, wqb, H_ * HW_ * DM_);
  cvt_bf16<<<(H_ * K_ * HW_ * DM_ + 255) / 256, 256, 0, stream>>>(Wk, wkb, H_ * K_ * HW_ * DM_);
  cvt_bf16<<<(H_ * K_ * HW_ * DM_ + 255) / 256, 256, 0, stream>>>(Wv, wvb, H_ * K_ * HW_ * DM_);
  cvt_bf16<<<(DM_ * DM_ + 255) / 256, 256, 0, stream>>>(Wo, wob, DM_ * DM_);

  dim3 g2(L_ / ML_, H_, B_);                        // 512 workgroups
  lcsa_qkv_attn<<<g2, 256, 0, stream>>>(xpad, wqb, wkb, wvb, bq, bk, bv, attnb);

  lcsa_outproj<<<(B_ * L_) / (16 * MT_), 256, 0, stream>>>(attnb, wob, bo, out);
}
